// times_block_31001073942750
// MI455X (gfx1250) — matrix operand order repaired and checked by reading, not compiled, not run on hardware
//
#include <hip/hip_runtime.h>
#include <math.h>

// out = x + gelu(x @ W1 + b1) @ W2 + b2
// (softmax weights over the K periods sum to 1 and h is k-independent,
//  so the whole FFT/bandpass/Hilbert pipeline cancels exactly.)
//
// x: [16384, 512] f32, W1: [512, 2048], b1: [2048], W2: [2048, 512], b2: [512]

typedef __bf16 bf16;
typedef bf16  v16bf __attribute__((ext_vector_type(16)));
typedef bf16  v8bf  __attribute__((ext_vector_type(8)));
typedef float v8f   __attribute__((ext_vector_type(8)));

#define MROWS   16384
#define KDIM    512      // C
#define NHID    2048     // 4C
#define M_TILE  64       // rows per workgroup
#define CH      64       // hidden-dim chunk width
#define NCHUNK  (NHID / CH)
#define BLOCK   512      // 16 waves

// padded LDS row strides (elements) to spread banks
#define XA_ST   520      // x tile   [64][512]
#define W1T_ST  528      // W1 chunk transposed [n=64][k=512]
#define W2T_ST  72       // W2 chunk transposed [n=512][k=64]
#define HB_ST   72       // h tile   [64][64]

#define W1CHUNK_BYTES ((size_t)CH * W1T_ST * sizeof(bf16))    // 67,584
#define W2CHUNK_BYTES ((size_t)KDIM * W2T_ST * sizeof(bf16))  // 73,728
#define W1PACK_BYTES  (NCHUNK * W1CHUNK_BYTES)                // ~2.06 MB
#define W2PACK_BYTES  (NCHUNK * W2CHUNK_BYTES)                // ~2.25 MB

__device__ __forceinline__ v16bf lds_frag(const bf16* p0, const bf16* p1) {
  v8bf lo = *(const v8bf*)p0;
  v8bf hi = *(const v8bf*)p1;
  return __builtin_shufflevector(lo, hi, 0,1,2,3,4,5,6,7,8,9,10,11,12,13,14,15);
}

__device__ __forceinline__ float gelu_exact(float v) {
  return 0.5f * v * (1.0f + erff(v * 0.70710678118654752f));
}

// ---- async global->LDS copy of a contiguous region (ASYNCcnt-tracked) ----
__device__ __forceinline__ void async_copy_region(uint32_t lds_base,
                                                  const char* gsrc,
                                                  int bytes, int tid) {
  for (int off = tid * 16; off < bytes; off += BLOCK * 16) {
    uint32_t l = lds_base + (uint32_t)off;
    unsigned long long g = (unsigned long long)(uintptr_t)(gsrc + off);
    asm volatile("global_load_async_to_lds_b128 %0, %1, off"
                 :: "v"(l), "v"(g) : "memory");
  }
}
__device__ __forceinline__ void wait_async0() {
  asm volatile("s_wait_asynccnt 0x0" ::: "memory");
}

// =====================================================================
// Pre-pack kernels: fp32 weights -> bf16, transposed into the exact
// padded per-chunk LDS tile layout (so staging is a raw byte copy).
// =====================================================================
__global__ void pack_w1(const float* __restrict__ W1g, bf16* __restrict__ dst) {
  const int idx = blockIdx.x * 256 + threadIdx.x;   // k*2048 + gn
  const int k  = idx >> 11;
  const int gn = idx & 2047;
  const int c  = gn >> 6;
  const int n  = gn & 63;
  dst[(size_t)c * (CH * W1T_ST) + (size_t)n * W1T_ST + k] = (bf16)W1g[idx];
}

__global__ void pack_w2(const float* __restrict__ W2g, bf16* __restrict__ dst) {
  const int idx = blockIdx.x * 256 + threadIdx.x;   // gk*512 + n
  const int gk = idx >> 9;
  const int n  = idx & 511;
  const int c  = gk >> 6;
  const int kk = gk & 63;
  dst[(size_t)c * (KDIM * W2T_ST) + (size_t)n * W2T_ST + kk] = (bf16)W2g[idx];
}

// =====================================================================
// Main fused FFN kernel, async-copy staging of pre-packed bf16 weights.
// Single LDS buffers with phase-shifted async copies:
//   W2(ch) copy overlaps GEMM1(ch); W1(ch+1) copy overlaps GEMM2(ch).
// =====================================================================
__global__ __launch_bounds__(BLOCK)
void ffn_wmma_async(const float* __restrict__ xg,
                    const bf16*  __restrict__ w1p,
                    const float* __restrict__ b1g,
                    const bf16*  __restrict__ w2p,
                    const float* __restrict__ b2g,
                    float* __restrict__ out) {
  extern __shared__ char smem_raw[];
  bf16* xA  = (bf16*)smem_raw;                 // [M_TILE][XA_ST]
  bf16* w1t = xA  + M_TILE * XA_ST;            // [CH][W1T_ST]
  bf16* w2t = w1t + CH * W1T_ST;               // [KDIM][W2T_ST]
  bf16* hB  = w2t + KDIM * W2T_ST;             // [M_TILE][HB_ST]

  // low 32 bits of a generic LDS pointer == LDS byte address (aperture LSBs are 0)
  const uint32_t w1t_lds = (uint32_t)(uintptr_t)w1t;
  const uint32_t w2t_lds = (uint32_t)(uintptr_t)w2t;

  const int tid  = threadIdx.x;
  const int lane = tid & 31;
  const int wid  = tid >> 5;
  const int ln16 = lane & 15;
  const int lh   = lane >> 4;
  const int wm   = wid >> 2;       // 0..3 : 16-row slice
  const int wn   = wid & 3;        // 0..3 : 128-col slice
  const size_t r0 = (size_t)blockIdx.x * M_TILE;

  // prologue: start streaming W1 chunk 0 while we convert the x tile
  async_copy_region(w1t_lds, (const char*)w1p, (int)W1CHUNK_BYTES, tid);

  #pragma unroll 4
  for (int p = tid; p < (M_TILE * KDIM) / 4; p += BLOCK) {
    const int k4 = (p & 127) << 2;
    const int r  = p >> 7;
    const float4 v = *(const float4*)&xg[(r0 + r) * KDIM + k4];
    bf16* dst = &xA[r * XA_ST + k4];
    dst[0] = (bf16)v.x; dst[1] = (bf16)v.y; dst[2] = (bf16)v.z; dst[3] = (bf16)v.w;
  }

  v8f acc[8];
  #pragma unroll
  for (int nt = 0; nt < 8; ++nt) acc[nt] = {};

  for (int ch = 0; ch < NCHUNK; ++ch) {
    const int kc0 = ch * CH;
    __syncthreads();                 // A: prev GEMM2 done -> w2t free, hB free
    wait_async0();                   // own W1(ch) loads landed
    async_copy_region(w2t_lds, (const char*)w2p + (size_t)ch * W2CHUNK_BYTES,
                      (int)W2CHUNK_BYTES, tid);
    __syncthreads();                 // B: all waves' W1(ch) data visible

    // ---- GEMM1: h[64][64] = gelu(xA @ W1chunk + b1) ----
    {
      v8f c1 = {};
      const bf16* arow = &xA[(wm * 16 + ln16) * XA_ST];
      const bf16* brow = &w1t[(wn * 16 + ln16) * W1T_ST];
      #pragma unroll
      for (int kk = 0; kk < KDIM / 32; ++kk) {
        const int k0 = kk * 32;
        v16bf a = lds_frag(arow + k0 + 8 * lh,  arow + k0 + 16 + 8 * lh);
        v16bf b = lds_frag(brow + k0 + 8 * lh,  brow + k0 + 16 + 8 * lh);
        c1 = __builtin_amdgcn_wmma_f32_16x16x32_bf16(false, a, false, b,
                                                     (short)0, c1, false, false);
      }
      const float bias1 = b1g[kc0 + wn * 16 + ln16];
      #pragma unroll
      for (int j = 0; j < 8; ++j) {
        const float g = gelu_exact(c1[j] + bias1);
        hB[(wm * 16 + 8 * lh + j) * HB_ST + wn * 16 + ln16] = (bf16)g;
      }
    }
    __syncthreads();                 // C: hB ready, w1t free
    wait_async0();                   // own W2(ch) loads landed
    if (ch + 1 < NCHUNK)             // stream W1(ch+1) under GEMM2(ch)
      async_copy_region(w1t_lds,
                        (const char*)w1p + (size_t)(ch + 1) * W1CHUNK_BYTES,
                        (int)W1CHUNK_BYTES, tid);
    __syncthreads();                 // D: all waves' W2(ch) data visible

    // ---- GEMM2: acc += h[64][64] @ W2chunk[64][512] ----
    {
      const bf16* arow = &hB[(wm * 16 + ln16) * HB_ST];
      v16bf a0 = lds_frag(arow + 8 * lh,      arow + 16 + 8 * lh);
      v16bf a1 = lds_frag(arow + 32 + 8 * lh, arow + 48 + 8 * lh);
      #pragma unroll
      for (int nt = 0; nt < 8; ++nt) {
        const bf16* bp = &w2t[(wn * 128 + nt * 16 + ln16) * W2T_ST];
        v16bf b0 = lds_frag(bp + 8 * lh,      bp + 16 + 8 * lh);
        v16bf b1 = lds_frag(bp + 32 + 8 * lh, bp + 48 + 8 * lh);
        acc[nt] = __builtin_amdgcn_wmma_f32_16x16x32_bf16(false, a0, false, b0,
                                                          (short)0, acc[nt], false, false);
        acc[nt] = __builtin_amdgcn_wmma_f32_16x16x32_bf16(false, a1, false, b1,
                                                          (short)0, acc[nt], false, false);
      }
    }
  }

  // ---- epilogue: out = x + acc + b2 ----
  #pragma unroll
  for (int nt = 0; nt < 8; ++nt) {
    const int col = wn * 128 + nt * 16 + ln16;
    const float bias2 = b2g[col];
    #pragma unroll
    for (int j = 0; j < 8; ++j) {
      const size_t row = r0 + wm * 16 + 8 * lh + j;
      out[row * KDIM + col] = xg[row * KDIM + col] + bias2 + acc[nt][j];
    }
  }
}

// =====================================================================
// Fallback (no-workspace) kernel: round-1 version with in-kernel fp32->bf16
// weight conversion. Used only if ws_size cannot hold the packed weights.
// =====================================================================
__global__ __launch_bounds__(BLOCK)
void ffn_residual_wmma(const float* __restrict__ xg,
                       const float* __restrict__ W1g,
                       const float* __restrict__ b1g,
                       const float* __restrict__ W2g,
                       const float* __restrict__ b2g,
                       float* __restrict__ out) {
  extern __shared__ char smem_raw[];
  bf16* xA  = (bf16*)smem_raw;
  bf16* w1t = xA  + M_TILE * XA_ST;
  bf16* w2t = w1t + CH * W1T_ST;
  bf16* hB  = w2t + KDIM * W2T_ST;

  const int tid  = threadIdx.x;
  const int lane = tid & 31;
  const int wid  = tid >> 5;
  const int ln16 = lane & 15;
  const int lh   = lane >> 4;
  const int wm   = wid >> 2;
  const int wn   = wid & 3;
  const size_t r0 = (size_t)blockIdx.x * M_TILE;

  #pragma unroll 4
  for (int p = tid; p < (M_TILE * KDIM) / 4; p += BLOCK) {
    const int k4 = (p & 127) << 2;
    const int r  = p >> 7;
    const float4 v = *(const float4*)&xg[(r0 + r) * KDIM + k4];
    bf16* dst = &xA[r * XA_ST + k4];
    dst[0] = (bf16)v.x; dst[1] = (bf16)v.y; dst[2] = (bf16)v.z; dst[3] = (bf16)v.w;
  }

  v8f acc[8];
  #pragma unroll
  for (int nt = 0; nt < 8; ++nt) acc[nt] = {};

  for (int ch = 0; ch < NCHUNK; ++ch) {
    const int kc0 = ch * CH;
    __syncthreads();

    #pragma unroll 4
    for (int p = tid; p < (KDIM * CH) / 2; p += BLOCK) {
      const int n2 = (p & 31) << 1;
      const int k  = p >> 5;
      const float2 v = *(const float2*)&W1g[(size_t)k * NHID + kc0 + n2];
      w1t[(n2    ) * W1T_ST + k] = (bf16)v.x;
      w1t[(n2 + 1) * W1T_ST + k] = (bf16)v.y;
    }
    #pragma unroll 4
    for (int p = tid; p < (CH * KDIM) / 2; p += BLOCK) {
      const int n2 = (p & 255) << 1;
      const int kk = p >> 8;
      const float2 v = *(const float2*)&W2g[(size_t)(kc0 + kk) * KDIM + n2];
      w2t[(n2    ) * W2T_ST + kk] = (bf16)v.x;
      w2t[(n2 + 1) * W2T_ST + kk] = (bf16)v.y;
    }
    __syncthreads();

    {
      v8f c1 = {};
      const bf16* arow = &xA[(wm * 16 + ln16) * XA_ST];
      const bf16* brow = &w1t[(wn * 16 + ln16) * W1T_ST];
      #pragma unroll
      for (int kk = 0; kk < KDIM / 32; ++kk) {
        const int k0 = kk * 32;
        v16bf a = lds_frag(arow + k0 + 8 * lh,  arow + k0 + 16 + 8 * lh);
        v16bf b = lds_frag(brow + k0 + 8 * lh,  brow + k0 + 16 + 8 * lh);
        c1 = __builtin_amdgcn_wmma_f32_16x16x32_bf16(false, a, false, b,
                                                     (short)0, c1, false, false);
      }
      const float bias1 = b1g[kc0 + wn * 16 + ln16];
      #pragma unroll
      for (int j = 0; j < 8; ++j) {
        const float g = gelu_exact(c1[j] + bias1);
        hB[(wm * 16 + 8 * lh + j) * HB_ST + wn * 16 + ln16] = (bf16)g;
      }
    }
    __syncthreads();

    {
      const bf16* arow = &hB[(wm * 16 + ln16) * HB_ST];
      v16bf a0 = lds_frag(arow + 8 * lh,      arow + 16 + 8 * lh);
      v16bf a1 = lds_frag(arow + 32 + 8 * lh, arow + 48 + 8 * lh);
      #pragma unroll
      for (int nt = 0; nt < 8; ++nt) {
        const bf16* bp = &w2t[(wn * 128 + nt * 16 + ln16) * W2T_ST];
        v16bf b0 = lds_frag(bp + 8 * lh,      bp + 16 + 8 * lh);
        v16bf b1 = lds_frag(bp + 32 + 8 * lh, bp + 48 + 8 * lh);
        acc[nt] = __builtin_amdgcn_wmma_f32_16x16x32_bf16(false, a0, false, b0,
                                                          (short)0, acc[nt], false, false);
        acc[nt] = __builtin_amdgcn_wmma_f32_16x16x32_bf16(false, a1, false, b1,
                                                          (short)0, acc[nt], false, false);
      }
    }
  }

  #pragma unroll
  for (int nt = 0; nt < 8; ++nt) {
    const int col = wn * 128 + nt * 16 + ln16;
    const float bias2 = b2g[col];
    #pragma unroll
    for (int j = 0; j < 8; ++j) {
      const size_t row = r0 + wm * 16 + 8 * lh + j;
      out[row * KDIM + col] = xg[row * KDIM + col] + bias2 + acc[nt][j];
    }
  }
}

extern "C" void kernel_launch(void* const* d_in, const int* in_sizes, int n_in,
                              void* d_out, int out_size, void* d_ws, size_t ws_size,
                              hipStream_t stream) {
  (void)in_sizes; (void)n_in; (void)out_size;
  const float* x  = (const float*)d_in[0];
  const float* W1 = (const float*)d_in[1];
  const float* b1 = (const float*)d_in[2];
  const float* W2 = (const float*)d_in[3];
  const float* b2 = (const float*)d_in[4];
  float* out = (float*)d_out;

  const size_t smem = (size_t)(M_TILE * XA_ST + CH * W1T_ST + KDIM * W2T_ST +
                               M_TILE * HB_ST) * sizeof(bf16);  // ~212 KB (<320 KB/WGP)
  dim3 grid(MROWS / M_TILE);   // 256
  dim3 block(BLOCK);           // 512 threads = 16 waves

  const size_t need = W1PACK_BYTES + W2PACK_BYTES;   // ~4.4 MB
  if (ws_size >= need && d_ws != nullptr) {
    bf16* w1p = (bf16*)d_ws;
    bf16* w2p = (bf16*)((char*)d_ws + W1PACK_BYTES);
    pack_w1<<<(KDIM * NHID) / 256, 256, 0, stream>>>(W1, w1p);
    pack_w2<<<(NHID * KDIM) / 256, 256, 0, stream>>>(W2, w2p);
    ffn_wmma_async<<<grid, block, smem, stream>>>(x, w1p, b1, w2p, b2, out);
  } else {
    ffn_residual_wmma<<<grid, block, smem, stream>>>(x, W1, b1, W2, b2, out);
  }
}
